// ChebNet_9766755631344
// MI455X (gfx1250) — compile-verified, matrix operand order checked by reading, not run on hardware
//
#include <hip/hip_runtime.h>
#include <hip/hip_bf16.h>

// ---------------------------------------------------------------------------
// ChebNet (K=5) for MI455X / gfx1250.
//   * edge norm + CSR-by-destination build (no fp32 atomics in hot loop)
//   * propagation = gather SpMM, fp32, L2-resident operands
//   * dense taps  = v_wmma_f32_16x16x32_bf16, A panel staged to LDS with
//     global_load_async_to_lds_b128 (ASYNCcnt) -- bf16 operand mirrors are
//     written by the producers so staging is a pure byte copy.
// ---------------------------------------------------------------------------

typedef __attribute__((ext_vector_type(16))) __bf16 v16bf;
typedef __attribute__((ext_vector_type(8)))  __bf16 v8bf;
typedef __attribute__((ext_vector_type(4)))  __bf16 v4bf;
typedef __attribute__((ext_vector_type(8)))  float  v8f;

__device__ __forceinline__ __bf16 f2bf(float f) { return (__bf16)f; }

// ------------------------------ graph prep --------------------------------

__global__ void k_deg(const int* __restrict__ row, const int* __restrict__ col,
                      float* __restrict__ deg, int E) {
  int e = blockIdx.x * blockDim.x + threadIdx.x;
  if (e >= E) return;
  int r = row[e], c = col[e];
  if (r != c) atomicAdd(&deg[r], 1.0f);
}

__global__ void k_dis(const float* __restrict__ deg, float* __restrict__ dis, int N) {
  int i = blockIdx.x * blockDim.x + threadIdx.x;
  if (i >= N) return;
  float d = deg[i];
  dis[i] = d > 0.f ? rsqrtf(d) : 0.f;
}

__global__ void k_w(const int* __restrict__ row, const int* __restrict__ col,
                    const float* __restrict__ dis, float* __restrict__ w, int E) {
  int e = blockIdx.x * blockDim.x + threadIdx.x;
  if (e >= E) return;
  int r = row[e], c = col[e];
  w[e] = (r != c) ? -dis[r] * dis[c] : 0.f;
}

__global__ void k_count(const int* __restrict__ col, unsigned* __restrict__ cnt, int E) {
  int e = blockIdx.x * blockDim.x + threadIdx.x;
  if (e >= E) return;
  atomicAdd(&cnt[col[e]], 1u);
}

// single-workgroup exclusive scan over N counters (N ~ 100k -> ~98 chunks)
__global__ void k_scan(const unsigned* __restrict__ cnt, unsigned* __restrict__ rowptr, int N) {
  __shared__ unsigned sh[1024];
  int t = threadIdx.x;
  unsigned carry = 0;
  int nchunks = (N + 1023) >> 10;
  for (int ch = 0; ch < nchunks; ++ch) {
    int i = (ch << 10) + t;
    unsigned v = (i < N) ? cnt[i] : 0u;
    sh[t] = v;
    __syncthreads();
    for (int off = 1; off < 1024; off <<= 1) {
      unsigned add = (t >= off) ? sh[t - off] : 0u;
      __syncthreads();
      sh[t] += add;
      __syncthreads();
    }
    if (i < N) rowptr[i] = carry + sh[t] - v;  // exclusive
    carry += sh[1023];
    __syncthreads();
  }
  if (t == 0) rowptr[N] = carry;
}

__global__ void k_scatter(const int* __restrict__ row, const int* __restrict__ col,
                          const float* __restrict__ w, const unsigned* __restrict__ rowptr,
                          unsigned* __restrict__ cursor, int* __restrict__ srcs,
                          float* __restrict__ wsort, int E) {
  int e = blockIdx.x * blockDim.x + threadIdx.x;
  if (e >= E) return;
  int c = col[e];
  unsigned p = rowptr[c] + atomicAdd(&cursor[c], 1u);
  srcs[p]  = row[e];
  wsort[p] = w[e];
}

// fp32 -> bf16 mirror (vectorized)
__global__ void k_cvt(const float* __restrict__ in, __bf16* __restrict__ out, long n4) {
  long i = (long)blockIdx.x * blockDim.x + threadIdx.x;
  if (i >= n4) return;
  float4 v = ((const float4*)in)[i];
  v4bf o;
  o[0] = f2bf(v.x); o[1] = f2bf(v.y); o[2] = f2bf(v.z); o[3] = f2bf(v.w);
  ((v4bf*)out)[i] = o;
}

// ------------------------------ propagation -------------------------------
// out[n][f] = alpha * sum_e w_e * X[src_e][f] + beta * Z[n][f]   (fp32)
// Also writes the bf16 mirror consumed by the following WMMA tap.
__global__ void k_prop(const float* __restrict__ X, const int* __restrict__ srcs,
                       const float* __restrict__ wsort, const unsigned* __restrict__ rowptr,
                       const float* __restrict__ Z, float* __restrict__ out,
                       __bf16* __restrict__ out_bf, float alpha, float beta) {
  int n = blockIdx.x;
  int f = threadIdx.x;  // 128
  unsigned beg = rowptr[n], end = rowptr[n + 1];
  __shared__ int   s_src[128];
  __shared__ float s_w[128];
  float acc = 0.f;
  for (unsigned base = beg; base < end; base += 128u) {
    unsigned m = end - base;
    if (m > 128u) m = 128u;
    if ((unsigned)f < m) {
      s_src[f] = srcs[base + f];
      s_w[f]   = wsort[base + f];
    }
    __syncthreads();
    for (unsigned j = 0; j < m; ++j)
      acc += s_w[j] * X[(long)s_src[j] * 128 + f];
    __syncthreads();
  }
  long o = (long)n * 128 + f;
  float v = alpha * acc + beta * Z[o];
  out[o] = v;
  out_bf[o] = f2bf(v);
}

// --------------------------- weight pre-packing ---------------------------
// Pack W[k][128][Freal] into exact WMMA B-fragment order (bf16):
// layout: [tap][kstep s][ntile t][lane][16 halves]
//   lane<16 : col = t*16+lane,    halves 0-7 -> K = s*32+0..7,   8-15 -> +16..23
//   lane>=16: col = t*16+lane-16, halves 0-7 -> K = s*32+8..15,  8-15 -> +24..31
__global__ void k_pack(const float* __restrict__ W, __bf16* __restrict__ dst,
                       int Freal, int ntiles, int total) {
  int idx = blockIdx.x * blockDim.x + threadIdx.x;
  if (idx >= total) return;
  int h    = idx & 15;
  int lane = (idx >> 4) & 31;
  int rem  = idx >> 9;
  int t    = rem % ntiles;  rem /= ntiles;
  int s    = rem & 3;
  int tap  = rem >> 2;
  int col  = t * 16 + (lane & 15);
  int hi   = lane >> 4;
  int k    = s * 32 + hi * 8 + (h & 7) + ((h >= 8) ? 16 : 0);
  float v  = (col < Freal) ? W[((long)tap * 128 + k) * Freal + col] : 0.f;
  dst[idx] = f2bf(v);
}

// ------------------------------- dense tap --------------------------------
// acc[N x ld] (+)= Xb[N x 128] @ Wtap.   256 threads = 8 waves; wave w owns
// rows [blk*128 + w*16, +16) and all NT column tiles. The full 128x128 bf16
// A panel (32 KB) is staged to LDS once per block with async-to-LDS ops.
// flags: bit0 = init C from bias, bit1 = ReLU epilogue, bit2 = write bf16
// mirror of the stored result (next layer's A operand).
template <int NT>
__global__ void __launch_bounds__(256) k_gemm(const __bf16* __restrict__ Xb,
                                              const __bf16* __restrict__ Bp,
                                              const float* __restrict__ bias,
                                              float* __restrict__ acc,
                                              __bf16* __restrict__ mirror,
                                              int n_rows, int ld, int nreal, int flags) {
  __shared__ __bf16 shA[128 * 128];
  int tid = threadIdx.x, lane = tid & 31, wave = tid >> 5;
  long row0  = (long)blockIdx.x * 128;
  long myrow = row0 + wave * 16;
  int nlo = lane & 15, hi = lane >> 4;
  bool fullM = (row0 + 128) <= (long)n_rows;

  // ---- async stage of the whole A panel (unguarded: mirrors have slack) ---
  const char* gsrc = (const char*)(Xb + row0 * 128);
#if __has_builtin(__builtin_amdgcn_global_load_async_to_lds_b128)
  {
    typedef int v4i_t __attribute__((ext_vector_type(4)));
    typedef __attribute__((address_space(1))) v4i_t* gp_t;
    typedef __attribute__((address_space(3))) v4i_t* lp_t;
    gp_t g = (gp_t)const_cast<char*>(gsrc);
    lp_t l = (lp_t)(void*)shA;
#pragma unroll
    for (int i = 0; i < 8; ++i) {
      int off = tid + i * 256;          // 16B vector units
      __builtin_amdgcn_global_load_async_to_lds_b128(g + off, l + off, 0, 0);
    }
  }
#else
#pragma unroll
  for (int i = 0; i < 8; ++i) {
    int off = tid + i * 256;
    ((int4*)shA)[off] = ((const int4*)gsrc)[off];
  }
#endif

  // ---- C init (overlaps the async copy) ----
  v8f c[NT];
  if (flags & 1) {
#pragma unroll
    for (int t = 0; t < NT; ++t) {
      int n = t * 16 + nlo;
      float bv = (n < nreal) ? bias[n] : 0.f;
#pragma unroll
      for (int r = 0; r < 8; ++r) c[t][r] = bv;
    }
  } else if (fullM) {
#pragma unroll
    for (int t = 0; t < NT; ++t) {
      int n = t * 16 + nlo;
      bool nok = n < nreal;
#pragma unroll
      for (int r = 0; r < 8; ++r) {
        long m = myrow + (hi ? 8 + r : r);
        c[t][r] = nok ? acc[m * ld + n] : 0.f;
      }
    }
  } else {
#pragma unroll
    for (int t = 0; t < NT; ++t) {
      int n = t * 16 + nlo;
#pragma unroll
      for (int r = 0; r < 8; ++r) {
        long m = myrow + (hi ? 8 + r : r);
        c[t][r] = (m < n_rows && n < nreal) ? acc[m * ld + n] : 0.f;
      }
    }
  }

#if __has_builtin(__builtin_amdgcn_global_load_async_to_lds_b128)
#if __has_builtin(__builtin_amdgcn_s_wait_asynccnt)
  __builtin_amdgcn_s_wait_asynccnt(0);
#else
  asm volatile("s_wait_asynccnt 0" ::: "memory");
#endif
#endif
  __syncthreads();

  // ---- K loop: 4 steps of 32, A fragment = two 16B LDS reads ----
  int rl = wave * 16 + nlo;
#pragma unroll
  for (int s = 0; s < 4; ++s) {
    const __bf16* ap = &shA[rl * 128 + s * 32 + hi * 8];
    v8bf a0 = *(const v8bf*)ap;
    v8bf a1 = *(const v8bf*)(ap + 16);
    v16bf a = __builtin_shufflevector(a0, a1, 0, 1, 2, 3, 4, 5, 6, 7,
                                      8, 9, 10, 11, 12, 13, 14, 15);
#pragma unroll
    for (int t = 0; t < NT; ++t) {
      v16bf b = *(const v16bf*)(Bp + (((long)s * NT + t) * 32 + lane) * 16);
      c[t] = __builtin_amdgcn_wmma_f32_16x16x32_bf16(false, a, false, b,
                                                     (short)0, c[t], false, false);
    }
  }

  // ---- epilogue ----
  bool relu = (flags & 2) != 0;
  bool wmir = (flags & 4) != 0;
#pragma unroll
  for (int t = 0; t < NT; ++t) {
    int n = t * 16 + nlo;
    if (n >= nreal) continue;
#pragma unroll
    for (int r = 0; r < 8; ++r) {
      long m = myrow + (hi ? 8 + r : r);
      if (fullM || m < n_rows) {
        float v = c[t][r];
        if (relu) v = fmaxf(v, 0.f);
        acc[m * ld + n] = v;
        if (wmir) mirror[m * 128 + n] = f2bf(v);
      }
    }
  }
}

// ------------------------------- launcher ---------------------------------

extern "C" void kernel_launch(void* const* d_in, const int* in_sizes, int n_in,
                              void* d_out, int out_size, void* d_ws, size_t ws_size,
                              hipStream_t stream) {
  const float* x  = (const float*)d_in[0];
  const float* W1 = (const float*)d_in[1];
  const float* b1 = (const float*)d_in[2];
  const float* W2 = (const float*)d_in[3];
  const float* b2 = (const float*)d_in[4];
  const float* W3 = (const float*)d_in[5];
  const float* b3 = (const float*)d_in[6];
  const int*   ei = (const int*)d_in[7];

  const int N    = in_sizes[0] / 128;
  const int E    = in_sizes[7] / 2;
  const int DOUT = in_sizes[6];            // 40
  const int* row = ei;
  const int* col = ei + E;

  // workspace carve (256B aligned)
  char* base = (char*)d_ws;
  size_t off = 0;
  auto carve = [&](size_t bytes) -> char* {
    char* q = base + off;
    off = (off + bytes + 255) & ~(size_t)255;
    return q;
  };
  float*    deg    = (float*)carve((size_t)N * 4);
  float*    dis    = (float*)carve((size_t)N * 4);
  float*    w      = (float*)carve((size_t)E * 4);
  unsigned* cnt    = (unsigned*)carve((size_t)N * 4);
  unsigned* rowptr = (unsigned*)carve((size_t)(N + 1) * 4);
  unsigned* cursor = (unsigned*)carve((size_t)N * 4);
  int*      srcs   = (int*)carve((size_t)E * 4);
  float*    wsort  = (float*)carve((size_t)E * 4);
  float*    T1 = (float*)carve((size_t)N * 128 * 4);
  float*    T2 = (float*)carve((size_t)N * 128 * 4);
  float*    T3 = (float*)carve((size_t)N * 128 * 4);
  float*    h1 = (float*)carve((size_t)N * 128 * 4);
  float*    h2 = (float*)carve((size_t)N * 128 * 4);
  // bf16 operand mirrors, over-allocated 32KB so tail blocks can stage
  // a full 128-row panel without per-element guards
  __bf16*   Xm = (__bf16*)carve((size_t)N * 128 * 2 + 32768);
  __bf16*   Tm = (__bf16*)carve((size_t)N * 128 * 2 + 32768);
  __bf16*   Wp1 = (__bf16*)carve((size_t)5 * 4 * 8 * 512 * 2);
  __bf16*   Wp2 = (__bf16*)carve((size_t)5 * 4 * 8 * 512 * 2);
  __bf16*   Wp3 = (__bf16*)carve((size_t)5 * 4 * 3 * 512 * 2);

  (void)hipMemsetAsync(deg, 0, (size_t)N * 4, stream);
  (void)hipMemsetAsync(cnt, 0, (size_t)N * 4, stream);
  (void)hipMemsetAsync(cursor, 0, (size_t)N * 4, stream);

  const int tb = 256;
  const int gE = (E + tb - 1) / tb;
  const int gN = (N + tb - 1) / tb;
  k_deg<<<gE, tb, 0, stream>>>(row, col, deg, E);
  k_dis<<<gN, tb, 0, stream>>>(deg, dis, N);
  k_w<<<gE, tb, 0, stream>>>(row, col, dis, w, E);
  k_count<<<gE, tb, 0, stream>>>(col, cnt, E);
  k_scan<<<1, 1024, 0, stream>>>(cnt, rowptr, N);
  k_scatter<<<gE, tb, 0, stream>>>(row, col, w, rowptr, cursor, srcs, wsort, E);

  {
    int tot = 5 * 4 * 8 * 512;
    k_pack<<<(tot + 255) / 256, 256, 0, stream>>>(W1, Wp1, 128, 8, tot);
    k_pack<<<(tot + 255) / 256, 256, 0, stream>>>(W2, Wp2, 128, 8, tot);
    int tot3 = 5 * 4 * 3 * 512;
    k_pack<<<(tot3 + 255) / 256, 256, 0, stream>>>(W3, Wp3, DOUT, 3, tot3);
  }

  // bf16 mirror of the network input (layer-1 tap-0 A operand)
  {
    long n4 = (long)N * 128 / 4;
    k_cvt<<<(int)((n4 + 255) / 256), 256, 0, stream>>>(x, Xm, n4);
  }

  const int gM = (N + 127) / 128;
  const long tapS8 = 4L * 8 * 512;
  const long tapS3 = 4L * 3 * 512;

  // hidden layers: F_out = 128, ReLU on last tap; last tap also emits the
  // bf16 mirror that becomes the next layer's tap-0 A operand.
  auto layer128 = [&](const float* Xf, const __bf16* Wp, const float* bias, float* out) {
    k_gemm<8><<<gM, 256, 0, stream>>>(Xm, Wp + 0 * tapS8, bias, out, (__bf16*)nullptr, N, 128, 128, 1);
    k_prop<<<N, 128, 0, stream>>>(Xf, srcs, wsort, rowptr, Xf, T1, Tm, 1.f, 0.f);   // T1 = L^ X
    k_gemm<8><<<gM, 256, 0, stream>>>(Tm, Wp + 1 * tapS8, bias, out, (__bf16*)nullptr, N, 128, 128, 0);
    k_prop<<<N, 128, 0, stream>>>(T1, srcs, wsort, rowptr, Xf, T2, Tm, 2.f, -1.f);  // T2 = 2L^T1 - X
    k_gemm<8><<<gM, 256, 0, stream>>>(Tm, Wp + 2 * tapS8, bias, out, (__bf16*)nullptr, N, 128, 128, 0);
    k_prop<<<N, 128, 0, stream>>>(T2, srcs, wsort, rowptr, T1, T3, Tm, 2.f, -1.f);  // T3 = 2L^T2 - T1
    k_gemm<8><<<gM, 256, 0, stream>>>(Tm, Wp + 3 * tapS8, bias, out, (__bf16*)nullptr, N, 128, 128, 0);
    k_prop<<<N, 128, 0, stream>>>(T3, srcs, wsort, rowptr, T2, T1, Tm, 2.f, -1.f);  // T4 (reuse T1)
    k_gemm<8><<<gM, 256, 0, stream>>>(Tm, Wp + 4 * tapS8, bias, out, Xm, N, 128, 128, 2 | 4);
  };
  layer128(x,  Wp1, b1, h1);
  layer128(h1, Wp2, b2, h2);

  // output layer: F_out = 40 (3 tiles padded to 48), no activation
  {
    float* o = (float*)d_out;
    k_gemm<3><<<gM, 256, 0, stream>>>(Xm, Wp3 + 0 * tapS3, b3, o, (__bf16*)nullptr, N, DOUT, DOUT, 1);
    k_prop<<<N, 128, 0, stream>>>(h2, srcs, wsort, rowptr, h2, T1, Tm, 1.f, 0.f);
    k_gemm<3><<<gM, 256, 0, stream>>>(Tm, Wp3 + 1 * tapS3, b3, o, (__bf16*)nullptr, N, DOUT, DOUT, 0);
    k_prop<<<N, 128, 0, stream>>>(T1, srcs, wsort, rowptr, h2, T2, Tm, 2.f, -1.f);
    k_gemm<3><<<gM, 256, 0, stream>>>(Tm, Wp3 + 2 * tapS3, b3, o, (__bf16*)nullptr, N, DOUT, DOUT, 0);
    k_prop<<<N, 128, 0, stream>>>(T2, srcs, wsort, rowptr, T1, T3, Tm, 2.f, -1.f);
    k_gemm<3><<<gM, 256, 0, stream>>>(Tm, Wp3 + 3 * tapS3, b3, o, (__bf16*)nullptr, N, DOUT, DOUT, 0);
    k_prop<<<N, 128, 0, stream>>>(T3, srcs, wsort, rowptr, T2, T1, Tm, 2.f, -1.f);
    k_gemm<3><<<gM, 256, 0, stream>>>(Tm, Wp3 + 4 * tapS3, b3, o, (__bf16*)nullptr, N, DOUT, DOUT, 0);
  }

  (void)n_in; (void)out_size; (void)ws_size;
}
